// MoEMLP_61297773248575
// MI455X (gfx1250) — matrix operand order repaired and checked by reading, not compiled, not run on hardware
//
#include <hip/hip_runtime.h>
#include <hip/hip_bf16.h>
#include <stdint.h>

#define N_TOK 8192
#define DDIM  1024
#define HDIM  4096
#define NEXP  8

#define MT 128                 // rows per tile
#define NT 256                 // output columns per workgroup (16 waves * 16)
#define KC 128                 // K chunk staged in LDS
#define AS_STRIDE (KC + 8)     // bf16 elems; keeps 16B align, uniform 2-way banks
#define ROWTILES (N_TOK / MT + NEXP)     // 72
#define NPAD (N_TOK + MT * NEXP)         // padded permuted rows

typedef __attribute__((ext_vector_type(16))) __bf16 v16bf;
typedef __attribute__((ext_vector_type(8)))  float  v8f;

union V16 { v16bf v; unsigned short s[16]; uint4 u[2]; };

// float -> bf16 bits, round-to-nearest-even
__device__ __forceinline__ unsigned short f2bf_bits(float f) {
    union { float f; unsigned u; } a; a.f = f;
    unsigned u = a.u;
    unsigned r = u + 0x7FFFu + ((u >> 16) & 1u);
    return (unsigned short)(r >> 16);
}
__device__ __forceinline__ unsigned pack_bf2(float lo, float hi) {
    return (unsigned)f2bf_bits(lo) | ((unsigned)f2bf_bits(hi) << 16);
}

// B-fragment: elems 0..7 = K[lanehalf*8 + i], elems 8..15 = K[16 + lanehalf*8 + i-8]
__device__ __forceinline__ v16bf load_b_bf16(const unsigned short* __restrict__ p) {
    V16 t;
    t.u[0] = *(const uint4*)(p);
    t.u[1] = *(const uint4*)(p + 16);
    return t.v;
}

// A-fragment from LDS row (bf16 row-major). klo = kbase + lanehalf*8.
// elems 0..7 = K[klo..klo+7], elems 8..15 = K[klo+16..klo+23] (ISA A16 layout)
__device__ __forceinline__ v16bf load_a_lds(const unsigned short* rowbase, int klo) {
    V16 t;
    t.u[0] = *(const uint4*)(rowbase + klo);
    t.u[1] = *(const uint4*)(rowbase + klo + 16);
    return t.v;
}

// ---------------- weight fp32 -> bf16 pre-conversion (memory-bound, one pass) ----
__global__ __launch_bounds__(256) void cvt_kernel(const float* __restrict__ src,
                                                  unsigned short* __restrict__ dst,
                                                  long n) {
    long i = ((long)blockIdx.x * 256 + (long)threadIdx.x) * 8;
    if (i < n) {
        const float4 a = *(const float4*)(src + i);
        const float4 b = *(const float4*)(src + i + 4);
        uint4 o;
        o.x = pack_bf2(a.x, a.y);
        o.y = pack_bf2(a.z, a.w);
        o.z = pack_bf2(b.x, b.y);
        o.w = pack_bf2(b.z, b.w);
        *(uint4*)(dst + i) = o;
    }
}

// ---------------- init: zero counts, perm = -1 ----------------
__global__ void init_kernel(int* counts, int* perm, int plen) {
    int i = blockIdx.x * blockDim.x + threadIdx.x;
    if (i < NEXP) counts[i] = 0;
    if (i < plen) perm[i] = -1;
}

// ---------------- router: top1 + histogram + x->bf16 ----------------
__global__ __launch_bounds__(256) void router_kernel(
    const float* __restrict__ x, const float* __restrict__ Wr, const float* __restrict__ br,
    int* __restrict__ top1, int* __restrict__ counts, unsigned short* __restrict__ xb)
{
    __shared__ float wr[NEXP * DDIM];   // 32 KB
    for (int i = threadIdx.x; i < NEXP * DDIM; i += 256) wr[i] = Wr[i];
    __syncthreads();

    const int lane = threadIdx.x & 31;
    const int wave = threadIdx.x >> 5;
    const int n = blockIdx.x * 8 + wave;             // one token per wave32
    const float* xrow = x + (size_t)n * DDIM;
    unsigned short* xbrow = xb + (size_t)n * DDIM;

    float acc[NEXP];
#pragma unroll
    for (int e = 0; e < NEXP; ++e) acc[e] = 0.f;

#pragma unroll
    for (int i = 0; i < DDIM / 128; ++i) {
        const int d = (i * 32 + lane) * 4;
        const float4 xv = *(const float4*)(xrow + d);
        uint2 pk;
        pk.x = pack_bf2(xv.x, xv.y);
        pk.y = pack_bf2(xv.z, xv.w);
        *(uint2*)(xbrow + d) = pk;
#pragma unroll
        for (int e = 0; e < NEXP; ++e) {
            const float4 wv = *(const float4*)(&wr[e * DDIM + d]);
            acc[e] += xv.x * wv.x + xv.y * wv.y + xv.z * wv.z + xv.w * wv.w;
        }
    }
#pragma unroll
    for (int e = 0; e < NEXP; ++e)
#pragma unroll
        for (int off = 16; off > 0; off >>= 1)
            acc[e] += __shfl_down(acc[e], off, 32);

    if (lane == 0) {
        int best = 0; float bv = acc[0] + br[0];
#pragma unroll
        for (int e = 1; e < NEXP; ++e) {
            float v = acc[e] + br[e];
            if (v > bv) { bv = v; best = e; }
        }
        top1[n] = best;
        atomicAdd(&counts[best], 1);
    }
}

// ---------------- scan: 128-aligned per-expert segments ----------------
__global__ void scan_kernel(const int* counts, int* tileb, int* cursor) {
    if (threadIdx.x == 0 && blockIdx.x == 0) {
        int tb = 0;
        for (int e = 0; e < NEXP; ++e) {
            tileb[e] = tb;
            cursor[e] = tb * MT;
            tb += (counts[e] + MT - 1) / MT;
        }
        tileb[NEXP] = tb;
    }
}

// ---------------- scatter tokens into expert-grouped list ----------------
__global__ void scatter_kernel(const int* __restrict__ top1, int* cursor, int* perm) {
    int n = blockIdx.x * blockDim.x + threadIdx.x;
    if (n < N_TOK) {
        int e = top1[n];
        int pos = atomicAdd(&cursor[e], 1);
        perm[pos] = n;
    }
}

// ---------------- tiled GEMM: 128 rows x 256 cols per 512-thread workgroup -----
// UP:  C = silu(A[perm] @ W^T + b)  -> hdst (bf16, padded-slot rows)
// DOWN:C = A @ W^T + b              -> outdst[perm] (f32)
template<bool UP, int KDIM, int NOUT>
__global__ __launch_bounds__(512) void gemm_kernel(
    const unsigned short* __restrict__ Asrc,   // UP: xb [N][D]; DOWN: hws [NPAD][H]
    const unsigned short* __restrict__ Wb,     // bf16 [E][NOUT][KDIM]
    const float* __restrict__ bias,            // [E][NOUT]
    unsigned short* __restrict__ hdst,         // UP only
    float* __restrict__ outdst,                // DOWN only
    const int* __restrict__ perm, const int* __restrict__ tileb)
{
    __shared__ unsigned short As[MT * AS_STRIDE];   // 128x128 bf16 chunk (~35 KB)
    __shared__ int toks[MT];

    const int bt = blockIdx.x;                      // row tile
    int e = -1;
#pragma unroll
    for (int i = 0; i < NEXP; ++i)
        if (bt >= tileb[i] && bt < tileb[i + 1]) e = i;
    if (e < 0) return;                              // uniform across block

    for (int t = threadIdx.x; t < MT; t += 512) toks[t] = perm[bt * MT + t];
    __syncthreads();

    const int lane = threadIdx.x & 31;
    const int wave = threadIdx.x >> 5;
    const int lm = lane & 15;                       // B/C column, A row within 16
    const int lh = lane >> 4;                       // lane half (K split)
    const int ncol = blockIdx.y * NT + wave * 16 + lm;
    const unsigned short* wrow =
        Wb + (size_t)e * NOUT * KDIM + (size_t)ncol * KDIM + lh * 8;

    // staging role: each thread owns 64B of one A row per chunk
    const int srow = threadIdx.x >> 2;              // 0..127
    const int quad = threadIdx.x & 3;
    const int stok = toks[srow];
    const bool sactive = (!UP) || (stok >= 0);
    const unsigned short* asrc =
        UP ? (stok >= 0 ? Asrc + (size_t)stok * KDIM : (const unsigned short*)0)
           : Asrc + ((size_t)bt * MT + srow) * KDIM;
    unsigned short* dstg = As + srow * AS_STRIDE + quad * 32;

    v8f acc[8];
#pragma unroll
    for (int rt = 0; rt < 8; ++rt) { v8f z = {}; acc[rt] = z; }

    // software pipeline: chunk kc+1 prefetched into regs while computing chunk kc
    uint4 q0 = {0u,0u,0u,0u}, q1 = q0, q2 = q0, q3 = q0;
    if (sactive) {
        const unsigned short* s = asrc + quad * 32;
        q0 = *(const uint4*)(s);
        q1 = *(const uint4*)(s + 8);
        q2 = *(const uint4*)(s + 16);
        q3 = *(const uint4*)(s + 24);
    }

    const int NCH = KDIM / KC;
    for (int kc = 0; kc < NCH; ++kc) {
        *(uint4*)(dstg)      = q0;
        *(uint4*)(dstg + 8)  = q1;
        *(uint4*)(dstg + 16) = q2;
        *(uint4*)(dstg + 24) = q3;
        __syncthreads();                            // chunk staged

        if (kc + 1 < NCH && sactive) {              // prefetch next chunk early
            const unsigned short* s = asrc + (kc + 1) * KC + quad * 32;
            q0 = *(const uint4*)(s);
            q1 = *(const uint4*)(s + 8);
            q2 = *(const uint4*)(s + 16);
            q3 = *(const uint4*)(s + 24);
        }

#pragma unroll
        for (int kk = 0; kk < KC / 32; ++kk) {
            const v16bf bfrag = load_b_bf16(wrow + kc * KC + kk * 32);
            const int klo = kk * 32 + lh * 8;
            // group-of-4 A fragments: keep several ds_loads in flight per WMMA burst
            v16bf a0 = load_a_lds(As + (0 * 16 + lm) * AS_STRIDE, klo);
            v16bf a1 = load_a_lds(As + (1 * 16 + lm) * AS_STRIDE, klo);
            v16bf a2 = load_a_lds(As + (2 * 16 + lm) * AS_STRIDE, klo);
            v16bf a3 = load_a_lds(As + (3 * 16 + lm) * AS_STRIDE, klo);
            acc[0] = __builtin_amdgcn_wmma_f32_16x16x32_bf16(
                false, a0, false, bfrag, (short)0, acc[0], false, false);
            acc[1] = __builtin_amdgcn_wmma_f32_16x16x32_bf16(
                false, a1, false, bfrag, (short)0, acc[1], false, false);
            acc[2] = __builtin_amdgcn_wmma_f32_16x16x32_bf16(
                false, a2, false, bfrag, (short)0, acc[2], false, false);
            acc[3] = __builtin_amdgcn_wmma_f32_16x16x32_bf16(
                false, a3, false, bfrag, (short)0, acc[3], false, false);
            a0 = load_a_lds(As + (4 * 16 + lm) * AS_STRIDE, klo);
            a1 = load_a_lds(As + (5 * 16 + lm) * AS_STRIDE, klo);
            a2 = load_a_lds(As + (6 * 16 + lm) * AS_STRIDE, klo);
            a3 = load_a_lds(As + (7 * 16 + lm) * AS_STRIDE, klo);
            acc[4] = __builtin_amdgcn_wmma_f32_16x16x32_bf16(
                false, a0, false, bfrag, (short)0, acc[4], false, false);
            acc[5] = __builtin_amdgcn_wmma_f32_16x16x32_bf16(
                false, a1, false, bfrag, (short)0, acc[5], false, false);
            acc[6] = __builtin_amdgcn_wmma_f32_16x16x32_bf16(
                false, a2, false, bfrag, (short)0, acc[6], false, false);
            acc[7] = __builtin_amdgcn_wmma_f32_16x16x32_bf16(
                false, a3, false, bfrag, (short)0, acc[7], false, false);
        }
        __syncthreads();                            // compute done, LDS reusable
    }

    // ---- epilogue
    const float bv = bias[(size_t)e * NOUT + ncol];
#pragma unroll
    for (int rt = 0; rt < 8; ++rt) {
#pragma unroll
        for (int r = 0; r < 8; ++r) {
            const int m = rt * 16 + lh * 8 + r;
            float v = acc[rt][r] + bv;
            if (UP) {
                v = v * __builtin_amdgcn_rcpf(1.f + __expf(-v));   // fast SiLU
                hdst[((size_t)(bt * MT + m)) * NOUT + ncol] = (unsigned short)f2bf_bits(v);
            } else {
                const int tok = toks[m];
                if (tok >= 0) outdst[(size_t)tok * NOUT + ncol] = v;
            }
        }
    }
}

extern "C" void kernel_launch(void* const* d_in, const int* in_sizes, int n_in,
                              void* d_out, int out_size, void* d_ws, size_t ws_size,
                              hipStream_t stream) {
    (void)in_sizes; (void)n_in; (void)out_size; (void)ws_size;
    const float* x     = (const float*)d_in[0];
    const float* Wr    = (const float*)d_in[1];
    const float* br    = (const float*)d_in[2];
    const float* Wup   = (const float*)d_in[3];
    const float* bup   = (const float*)d_in[4];
    const float* Wdown = (const float*)d_in[5];
    const float* bdown = (const float*)d_in[6];
    float* out = (float*)d_out;

    // workspace layout
    int* top1   = (int*)d_ws;                 // N
    int* counts = top1 + N_TOK;               // E
    int* tileb  = counts + NEXP;              // E+1
    int* cursor = tileb + (NEXP + 1);         // E
    int* perm   = cursor + NEXP;              // NPAD
    uintptr_t p = (uintptr_t)(perm + NPAD);
    p = (p + 255) & ~(uintptr_t)255;
    unsigned short* xb  = (unsigned short*)p;               // N*D bf16    (~16 MB)
    unsigned short* wub = xb  + (size_t)N_TOK * DDIM;       // E*H*D bf16  (~64 MB)
    unsigned short* wdb = wub + (size_t)NEXP * HDIM * DDIM; // E*D*H bf16  (~64 MB)
    unsigned short* hws = wdb + (size_t)NEXP * HDIM * DDIM; // NPAD*H bf16 (~76 MB)

    const long WN = (long)NEXP * HDIM * DDIM;               // 33.5M elems each

    init_kernel<<<(NPAD + 255) / 256, 256, 0, stream>>>(counts, perm, NPAD);
    cvt_kernel<<<(int)(WN / (256 * 8)), 256, 0, stream>>>(Wup, wub, WN);
    cvt_kernel<<<(int)(WN / (256 * 8)), 256, 0, stream>>>(Wdown, wdb, WN);
    router_kernel<<<N_TOK / 8, 256, 0, stream>>>(x, Wr, br, top1, counts, xb);
    scan_kernel<<<1, 32, 0, stream>>>(counts, tileb, cursor);
    scatter_kernel<<<N_TOK / 256, 256, 0, stream>>>(top1, cursor, perm);

    gemm_kernel<true, DDIM, HDIM>
        <<<dim3(ROWTILES, HDIM / NT), 512, 0, stream>>>(
            xb, wub, bup, hws, nullptr, perm, tileb);
    gemm_kernel<false, HDIM, DDIM>
        <<<dim3(ROWTILES, DDIM / NT), 512, 0, stream>>>(
            hws, wdb, bdown, nullptr, out, perm, tileb);
}
